// SpatialPart_27462020890953
// MI455X (gfx1250) — matrix operand order repaired and checked by reading, not compiled, not run on hardware
//
#include <hip/hip_runtime.h>
#include <hip/hip_bf16.h>

// ---------------------------------------------------------------------------
// Channel-attention stack for MI455X (gfx1250), bf16 WMMA path, v2.
//   x:(512,2,2048,17) -> h:(34,2048) per batch -> 3 attention layers -> head
// One workgroup (256 threads = 8 wave32) per batch element.
// h kept TRANSPOSED in LDS (sHT[t][c], bf16) so every WMMA B-operand is
// 8 contiguous dword LDS loads; D-tiles write back as packed b32 stores with
// native f32->bf16 conversion. All GEMMs via v_wmma_f32_16x16x32_bf16.
// ---------------------------------------------------------------------------

#define T_LEN  2048
#define HTS    52              // sHT row stride (elements): 48 cols c + pad
#define SS     36              // Q/K staging row stride
#define NWAVES 8
#define NCHUNK (T_LEN / 32)    // 64 T-chunks of 32

typedef __attribute__((ext_vector_type(16))) __bf16 v16bf;
typedef __attribute__((ext_vector_type(8)))  float  v8f;

union BFTile { v16bf v; unsigned short h[16]; unsigned u[8]; };

// LDS element budget (u16)
#define SHT_E    (T_LEN * HTS)          // h^T  (t-major), bf16
#define SW_E     (3 * 48 * 64)          // padded Wq/Wk/Wv
#define SATTN_E  (48 * 64)              // padded attn
#define SSTAGE_E (NWAVES * 2 * 48 * SS) // per-wave Q/K (or V^T) staging
#define U16_TOTAL (SHT_E + SW_E + SATTN_E + SSTAGE_E)
#define F32_TOTAL (48 * 48 + 3 * 48 + 660)
#define SMEM_BYTES ((size_t)U16_TOTAL * 2 + (size_t)F32_TOTAL * 4)

// head-constants layout inside sHead (floats)
#define HW1   0      // 16x34
#define HB1   544    // 16
#define HW2   560    // 3x16
#define HB2   608    // 3 (pad to 4)
#define HMEAN 612    // 16
#define HINV  628    // 16
#define HBETA 644    // 16

// native f32 -> bf16 (hardware cvt, RNE)
__device__ __forceinline__ unsigned short f2bf(float f) {
  union { __bf16 b; unsigned short u; } x;
  x.b = (__bf16)f;
  return x.u;
}
__device__ __forceinline__ unsigned pack2bf(float lo, float hi) {
  union { __bf16 b[2]; unsigned u; } x;
  x.b[0] = (__bf16)lo;
  x.b[1] = (__bf16)hi;
  return x.u;
}
__device__ __forceinline__ float bf2f(unsigned short h) {
  union { unsigned u; float f; } x; x.u = ((unsigned)h) << 16;
  return x.f;
}

__device__ __forceinline__ v8f v8f_zero() {
  v8f z;
#pragma unroll
  for (int i = 0; i < 8; ++i) z[i] = 0.0f;
  return z;
}

__device__ __forceinline__ v8f wmma_bf16(v16bf a, v16bf b, v8f c) {
  return __builtin_amdgcn_wmma_f32_16x16x32_bf16(
      /*neg_a=*/false, a, /*neg_b=*/false, b,
      /*c_mod=*/(short)0, c, /*reuse_a=*/false, /*reuse_b=*/false);
}

// A operand: 16(M) x 32(K), source buf row-major buf[m][k].
// Lane l: m=l&15, half=l>>4. VGPR j: K = 2j + 16*half.
__device__ __forceinline__ v16bf loadA(const unsigned short* buf, int rs,
                                       int r0, int c0) {
  BFTile t;
  const int lane = threadIdx.x & 31;
  const int m = lane & 15, half = lane >> 4;
  const unsigned short* p = buf + (r0 + m) * rs + c0 + half * 16;
#pragma unroll
  for (int j = 0; j < 8; ++j) t.u[j] = *(const unsigned*)(p + 2 * j);
  return t.v;
}

// B operand: 32(K) x 16(N), source holds B^T row-major: buf[n][k].
// Lane l: n=l&15, half=l>>4. VGPR j: K = 2j + 16*half.  8 dword loads.
__device__ __forceinline__ v16bf loadBT(const unsigned short* buf, int rs,
                                        int n0, int k0) {
  BFTile t;
  const int lane = threadIdx.x & 31;
  const int n = lane & 15, half = lane >> 4;
  const unsigned short* p = buf + (n0 + n) * rs + k0 + 16 * half;
#pragma unroll
  for (int j = 0; j < 8; ++j) t.u[j] = *(const unsigned*)(p + 2 * j);
  return t.v;
}

// Same, but upper lane-half (K=k0+16..k0+31) is all zero (K-dim padding):
// used for the second K-step where only K=k0..k0+15 exists in the buffer.
__device__ __forceinline__ v16bf loadBT_hi0(const unsigned short* buf, int rs,
                                            int n0, int k0) {
  BFTile t;
  const int lane = threadIdx.x & 31;
  const int n = lane & 15, half = lane >> 4;
  if (half) {
#pragma unroll
    for (int j = 0; j < 8; ++j) t.u[j] = 0u;
  } else {
    const unsigned short* p = buf + (n0 + n) * rs + k0;
#pragma unroll
    for (int j = 0; j < 8; ++j) t.u[j] = *(const unsigned*)(p + 2 * j);
  }
  return t.v;
}

__global__ void __launch_bounds__(256)
spatial_attn_kernel(const float* __restrict__ x,
                    const float* __restrict__ wq, const float* __restrict__ bq,
                    const float* __restrict__ wk, const float* __restrict__ bk,
                    const float* __restrict__ wv, const float* __restrict__ bv,
                    const float* __restrict__ w1, const float* __restrict__ b1,
                    const float* __restrict__ gamma, const float* __restrict__ beta,
                    const float* __restrict__ mean, const float* __restrict__ var,
                    const float* __restrict__ w2, const float* __restrict__ b2,
                    float* __restrict__ out) {
  extern __shared__ unsigned char smem[];
  unsigned short* sHT    = (unsigned short*)smem;        // T_LEN x HTS (h^T)
  unsigned short* sW     = sHT + SHT_E;                  // 3 x 48 x 64
  unsigned short* sAttn  = sW + SW_E;                    // 48 x 64
  unsigned short* sStage = sAttn + SATTN_E;              // 8 x 2 x 48 x SS
  float* sScores = (float*)(sStage + SSTAGE_E);          // 48 x 48
  float* sBias   = sScores + 48 * 48;                    // 3 x 48
  float* sHead   = sBias + 3 * 48;                       // 660

  const int b     = blockIdx.x;
  const int tid   = threadIdx.x;
  const int lane  = tid & 31;
  const int wave  = tid >> 5;
  const int dcol  = lane & 15;   // D-tile column for this lane
  const int dhalf = lane >> 4;   // D-tile row-half

  // ---- build h^T (2048 x 34, bf16) from x(b, 2, 2048, 17); pad cols 34..47 --
  const float* xb = x + (size_t)b * 2 * T_LEN * 17;
  for (int e = tid; e < 2 * T_LEN * 17; e += 256) {
    int v = e % 17;
    int t = (e / 17) % T_LEN;
    int c = e / (17 * T_LEN);
    sHT[t * HTS + c * 17 + v] = f2bf(xb[e]);
  }
  for (int e = tid; e < 14 * T_LEN; e += 256) {
    int t = e % T_LEN;
    int c = 34 + e / T_LEN;
    sHT[t * HTS + c] = 0;
  }
  // ---- head constants into LDS ----
  for (int e = tid; e < 544; e += 256) sHead[HW1 + e] = w1[e];
  if (tid < 16) {
    sHead[HB1 + tid]   = b1[tid];
    sHead[HMEAN + tid] = mean[tid];
    sHead[HINV + tid]  = gamma[tid] * rsqrtf(var[tid] + 1e-5f);
    sHead[HBETA + tid] = beta[tid];
  }
  if (tid < 48) sHead[HW2 + tid] = w2[tid];
  if (tid < 3)  sHead[HB2 + tid] = b2[tid];

  const float scale = rsqrtf((float)T_LEN);
  unsigned short* qbuf = sStage + wave * 2 * 48 * SS;  // Q staging [c][t], 48xSS
  unsigned short* kbuf = qbuf + 48 * SS;               // K staging [d][t], 48xSS
  unsigned short* vbufT = qbuf;                        // V^T staging [t][d], 32xHTS (reuse)

  for (int layer = 0; layer < 3; ++layer) {
    __syncthreads();
    // ---- stage padded weights (48x64 bf16) + biases, zero scores ----
    for (int e = tid; e < 3 * 48 * 64; e += 256) {
      int mat = e / 3072;
      int rem = e - mat * 3072;
      int r = rem >> 6, k = rem & 63;
      float v = 0.0f;
      if (r < 34 && k < 34) {
        const float* w = (mat == 0) ? wq : (mat == 1) ? wk : wv;
        v = w[layer * 1156 + r * 34 + k];
      }
      sW[e] = f2bf(v);
    }
    for (int e = tid; e < 3 * 48; e += 256) {
      int mat = e / 48, r = e % 48;
      const float* bb = (mat == 0) ? bq : (mat == 1) ? bk : bv;
      sBias[e] = (r < 34) ? bb[layer * 34 + r] : 0.0f;
    }
    for (int e = tid; e < 48 * 48; e += 256) sScores[e] = 0.0f;
    __syncthreads();

    // ---- pass 1: scores = (Wq h + bq)(Wk h + bk)^T, accumulated over T ----
    v8f acc[3][3];
#pragma unroll
    for (int i = 0; i < 3; ++i)
#pragma unroll
      for (int j = 0; j < 3; ++j) acc[i][j] = v8f_zero();

    for (int ch = wave; ch < NCHUNK; ch += NWAVES) {
      const int tbase = ch * 32;
      // Q and K for this 34x32 chunk into per-wave staging (row-major [c][t])
#pragma unroll
      for (int sel = 0; sel < 2; ++sel) {
        unsigned short* dst = sel ? kbuf : qbuf;
        const unsigned short* Wm = sW + sel * 3072;
#pragma unroll
        for (int nt = 0; nt < 2; ++nt) {
          v16bf b0 = loadBT(sHT, HTS, tbase + nt * 16, 0);
          v16bf b1t = loadBT_hi0(sHT, HTS, tbase + nt * 16, 32);
#pragma unroll
          for (int mt = 0; mt < 3; ++mt) {
            v8f d = wmma_bf16(loadA(Wm, 64, mt * 16, 0), b0, v8f_zero());
            d = wmma_bf16(loadA(Wm, 64, mt * 16, 32), b1t, d);
            const int base = mt * 16 + 8 * dhalf;
#pragma unroll
            for (int r = 0; r < 8; ++r)
              dst[(base + r) * SS + nt * 16 + dcol] =
                  f2bf(d[r] + sBias[sel * 48 + base + r]);
          }
        }
      }
      // scores += Q_chunk @ K_chunk^T  (K-dim = 32 t-values)
      v16bf qa[3], kb[3];
#pragma unroll
      for (int i = 0; i < 3; ++i) qa[i] = loadA(qbuf, SS, i * 16, 0);
#pragma unroll
      for (int i = 0; i < 3; ++i) kb[i] = loadBT(kbuf, SS, i * 16, 0);
#pragma unroll
      for (int mt = 0; mt < 3; ++mt)
#pragma unroll
        for (int nt = 0; nt < 3; ++nt)
          acc[mt][nt] = wmma_bf16(qa[mt], kb[nt], acc[mt][nt]);
    }
    // cross-wave reduction of score tiles
#pragma unroll
    for (int mt = 0; mt < 3; ++mt)
#pragma unroll
      for (int nt = 0; nt < 3; ++nt)
#pragma unroll
        for (int r = 0; r < 8; ++r)
          atomicAdd(&sScores[(mt * 16 + r + 8 * dhalf) * 48 + nt * 16 + dcol],
                    acc[mt][nt][r] * scale);
    __syncthreads();

    // ---- softmax over rows (34x34) -> padded bf16 attn 48x64 ----
    if (tid < 48) {
      int r = tid;
      if (r < 34) {
        float mx = -1e30f;
        for (int d2 = 0; d2 < 34; ++d2) mx = fmaxf(mx, sScores[r * 48 + d2]);
        float s = 0.0f;
        for (int d2 = 0; d2 < 34; ++d2) s += __expf(sScores[r * 48 + d2] - mx);
        float inv = 1.0f / s;
        for (int d2 = 0; d2 < 34; ++d2)
          sAttn[r * 64 + d2] = f2bf(__expf(sScores[r * 48 + d2] - mx) * inv);
        for (int d2 = 34; d2 < 64; ++d2) sAttn[r * 64 + d2] = 0;
      } else {
        for (int d2 = 0; d2 < 64; ++d2) sAttn[r * 64 + d2] = 0;
      }
    }
    __syncthreads();

    // ---- pass 2: V chunk, then h <- attn @ V, in place per wave-owned chunk --
    // attn A-tiles are chunk-invariant: hoist into registers (6 x v16bf).
    v16bf attnA[3][2];
#pragma unroll
    for (int mt = 0; mt < 3; ++mt)
#pragma unroll
      for (int ks = 0; ks < 2; ++ks)
        attnA[mt][ks] = loadA(sAttn, 64, mt * 16, ks * 32);

    for (int ch = wave; ch < NCHUNK; ch += NWAVES) {
      const int tbase = ch * 32;
      const unsigned short* Wm = sW + 2 * 3072;
      // V^T staging [t][d] with packed bf16 pair stores
#pragma unroll
      for (int nt = 0; nt < 2; ++nt) {
        v16bf b0 = loadBT(sHT, HTS, tbase + nt * 16, 0);
        v16bf b1t = loadBT_hi0(sHT, HTS, tbase + nt * 16, 32);
#pragma unroll
        for (int mt = 0; mt < 3; ++mt) {
          v8f d = wmma_bf16(loadA(Wm, 64, mt * 16, 0), b0, v8f_zero());
          d = wmma_bf16(loadA(Wm, 64, mt * 16, 32), b1t, d);
          const int base = mt * 16 + 8 * dhalf;
          unsigned* p = (unsigned*)(vbufT + (nt * 16 + dcol) * HTS + base);
#pragma unroll
          for (int r = 0; r < 8; r += 2)
            p[r >> 1] = pack2bf(d[r] + sBias[96 + base + r],
                                d[r + 1] + sBias[96 + base + r + 1]);
        }
      }
      // h^T[:, chunk] <- attn @ V (packed writeback)
#pragma unroll
      for (int nt = 0; nt < 2; ++nt) {
        v16bf b0 = loadBT(vbufT, HTS, nt * 16, 0);
        v16bf b1t = loadBT_hi0(vbufT, HTS, nt * 16, 32);
#pragma unroll
        for (int mt = 0; mt < 3; ++mt) {
          v8f d = wmma_bf16(attnA[mt][0], b0, v8f_zero());
          d = wmma_bf16(attnA[mt][1], b1t, d);
          const int base = mt * 16 + 8 * dhalf;
          unsigned* p = (unsigned*)(sHT + (tbase + nt * 16 + dcol) * HTS + base);
#pragma unroll
          for (int r = 0; r < 8; r += 2)
            p[r >> 1] = pack2bf(d[r], d[r + 1]);
        }
      }
    }
    __syncthreads();
  }

  // ---- head: y = LeakyReLU(BN(w1 h + b1)); out = w2 y + b2 (scalar tail) ----
  for (int t = tid; t < T_LEN; t += 256) {
    float hcol[34];
#pragma unroll
    for (int c = 0; c < 34; ++c) hcol[c] = bf2f(sHT[t * HTS + c]);
    float yv[16];
#pragma unroll
    for (int m = 0; m < 16; ++m) {
      float s = sHead[HB1 + m];
      const float* wr = sHead + HW1 + m * 34;
#pragma unroll
      for (int c = 0; c < 34; ++c) s += wr[c] * hcol[c];
      s = (s - sHead[HMEAN + m]) * sHead[HINV + m] + sHead[HBETA + m];
      yv[m] = (s > 0.0f) ? s : 0.01f * s;
    }
#pragma unroll
    for (int o = 0; o < 3; ++o) {
      float s = sHead[HB2 + o];
      const float* wr = sHead + HW2 + o * 16;
#pragma unroll
      for (int m = 0; m < 16; ++m) s += wr[m] * yv[m];
      out[((size_t)b * 3 + o) * T_LEN + t] = s;
    }
  }
}

extern "C" void kernel_launch(void* const* d_in, const int* in_sizes, int n_in,
                              void* d_out, int out_size, void* d_ws, size_t ws_size,
                              hipStream_t stream) {
  const float* x     = (const float*)d_in[0];
  const float* wq    = (const float*)d_in[1];
  const float* bq    = (const float*)d_in[2];
  const float* wk    = (const float*)d_in[3];
  const float* bk    = (const float*)d_in[4];
  const float* wv    = (const float*)d_in[5];
  const float* bv    = (const float*)d_in[6];
  const float* w1    = (const float*)d_in[7];
  const float* b1    = (const float*)d_in[8];
  const float* gamma = (const float*)d_in[9];
  const float* beta  = (const float*)d_in[10];
  const float* mean  = (const float*)d_in[11];
  const float* var   = (const float*)d_in[12];
  const float* w2    = (const float*)d_in[13];
  const float* b2    = (const float*)d_in[14];
  float* out = (float*)d_out;

  const int B = in_sizes[0] / (2 * T_LEN * 17);  // 512

  hipFuncSetAttribute(reinterpret_cast<const void*>(spatial_attn_kernel),
                      hipFuncAttributeMaxDynamicSharedMemorySize,
                      (int)SMEM_BYTES);
  spatial_attn_kernel<<<B, 256, SMEM_BYTES, stream>>>(
      x, wq, bq, wk, bk, wv, bv, w1, b1, gamma, beta, mean, var, w2, b2, out);
}
